// PredictiveCodingNetwork_65274912965006
// MI455X (gfx1250) — matrix operand order repaired and checked by reading, not compiled, not run on hardware
//
#include <hip/hip_runtime.h>
#include <hip/hip_bf16.h>
#include <stdint.h>

#define N_PATCH 4
#define N_SETTLE 3
#define D_MODEL 2048
#define D_REPR 128
#define BATCH 4
#define SEQ 2048

typedef __attribute__((ext_vector_type(16))) __bf16 v16bf;
typedef __attribute__((ext_vector_type(8)))  __bf16 v8bf;
typedef __attribute__((ext_vector_type(8)))  float  v8f;

// ---------------------------------------------------------------------------
// WMMA fragment helpers (16x16x32 bf16, f32 accumulate), per CDNA5 ISA layouts
// A (16x32, 16-bit): lane l<16 -> row l, K = {0..7, 16..23}; lane>=16 -> +8
// B (32x16, 16-bit): lane l<16 -> col l, K = {0..7, 16..23}; lane>=16 -> col l-16, +8
// C/D (16x16 f32): vgpr r -> row r (lanes 0-15) / row 8+r (lanes 16-31), col = lane&15
// ---------------------------------------------------------------------------

__device__ __forceinline__ v16bf ld_a_f32(const float* __restrict__ p) {
  float4 x0 = *reinterpret_cast<const float4*>(p);
  float4 x1 = *reinterpret_cast<const float4*>(p + 4);
  float4 y0 = *reinterpret_cast<const float4*>(p + 16);
  float4 y1 = *reinterpret_cast<const float4*>(p + 20);
  v16bf a;
  a[0]=(__bf16)x0.x; a[1]=(__bf16)x0.y; a[2]=(__bf16)x0.z; a[3]=(__bf16)x0.w;
  a[4]=(__bf16)x1.x; a[5]=(__bf16)x1.y; a[6]=(__bf16)x1.z; a[7]=(__bf16)x1.w;
  a[8]=(__bf16)y0.x; a[9]=(__bf16)y0.y; a[10]=(__bf16)y0.z; a[11]=(__bf16)y0.w;
  a[12]=(__bf16)y1.x; a[13]=(__bf16)y1.y; a[14]=(__bf16)y1.z; a[15]=(__bf16)y1.w;
  return a;
}

__device__ __forceinline__ v16bf ld_a_bf(const __bf16* __restrict__ p) {
  v8bf lo = *reinterpret_cast<const v8bf*>(p);
  v8bf hi = *reinterpret_cast<const v8bf*>(p + 16);
  v16bf a;
#pragma unroll
  for (int j = 0; j < 8; ++j) { a[j] = lo[j]; a[8 + j] = hi[j]; }
  return a;
}

__device__ __forceinline__ v16bf ld_b(const __bf16* __restrict__ p) {
  return ld_a_bf(p);  // col n, K = {0..7, 16..23} (+8 for lanes>=16)
}

__device__ __forceinline__ void st_c(float* __restrict__ O, int ld, int m, int nbase,
                                     int lane, v8f acc, const float* __restrict__ bias) {
  int n = nbase + (lane & 15);
  int ro = (lane & 16) ? 8 : 0;
  float bv = bias[n];
#pragma unroll
  for (int r = 0; r < 8; ++r) O[(size_t)(m + ro + r) * ld + n] = acc[r] + bv;
}

#define WMMA_BF16(a, b, c) \
  __builtin_amdgcn_wmma_f32_16x16x32_bf16(false, (a), false, (b), (short)0, (c), false, false)

// ---------------------------------------------------------------------------
// Weight prep: transpose + convert to bf16 [n][k] layout for B operands
// ---------------------------------------------------------------------------
__global__ void pcn_prep_wct(const float* __restrict__ Wc, __bf16* __restrict__ WcT) {
  size_t idx = (size_t)blockIdx.x * 256 + threadIdx.x;           // 4*128*2048
  int k = (int)(idx % D_MODEL);
  size_t t = idx / D_MODEL;
  int n = (int)(t % D_REPR);
  int p = (int)(t / D_REPR);
  WcT[idx] = (__bf16)Wc[((size_t)p * D_MODEL + k) * D_REPR + n];
}

__global__ void pcn_prep_wsum(const float* __restrict__ Wprec, __bf16* __restrict__ WsumT) {
  size_t idx = (size_t)blockIdx.x * 256 + threadIdx.x;           // 4*128*128
  int k = (int)(idx % D_REPR);
  int n = (int)((idx / D_REPR) % D_REPR);
  int p = (int)(idx / (D_REPR * D_REPR));
  const float* W = Wprec + (size_t)p * 2 * D_REPR * D_REPR;
  WsumT[idx] = (__bf16)(W[(size_t)k * D_REPR + n] + W[(size_t)(D_REPR + k) * D_REPR + n]);
}

__global__ void pcn_prep_wcorr(const float* __restrict__ Wcorr, __bf16* __restrict__ WcorrT) {
  size_t idx = (size_t)blockIdx.x * 256 + threadIdx.x;           // 4*2048*128
  int k = (int)(idx % D_REPR);
  size_t t = idx / D_REPR;
  int n = (int)(t % D_MODEL);
  int p = (int)(t / D_MODEL);
  WcorrT[idx] = (__bf16)Wcorr[((size_t)p * D_REPR + k) * D_MODEL + n];
}

__global__ void pcn_zero_vec(float* __restrict__ pred, float* __restrict__ predWb) {
  int i = blockIdx.x * 256 + threadIdx.x;
  if (i < N_PATCH * BATCH * D_REPR) { pred[i] = 0.f; predWb[i] = 0.f; }
}

// ---------------------------------------------------------------------------
// GEMM (M-tile 128, N=128, runtime K): used for C = h@Wc + bc  and  Zc = C@Wsum + bprec
// grid: (seqtile=16, batch=4, patch=4), 256 threads = 8 waves, wave w -> rows 16w..16w+15
// ---------------------------------------------------------------------------
__global__ __launch_bounds__(256) void pcn_gemm_n128(const float* __restrict__ A,
                                                     const __bf16* __restrict__ Bt,
                                                     const float* __restrict__ bias,
                                                     float* __restrict__ Out, int K) {
  int st = blockIdx.x, b = blockIdx.y, p = blockIdx.z;
  int lane = threadIdx.x & 31, wave = threadIdx.x >> 5;
  int pb = p * BATCH + b;
  const float*  Ab = A  + ((size_t)pb * SEQ + st * 128) * (size_t)K;
  const __bf16* Bb = Bt + (size_t)p * D_REPR * (size_t)K;
  const float*  bs = bias + (size_t)p * D_REPR;

  int m = wave * 16;
  int arow  = m + (lane & 15);
  int akoff = (lane & 16) ? 8 : 0;
  int bn    = lane & 15;
  int bkoff = (lane & 16) ? 8 : 0;

  v8f acc[8] = {};
  const float* pa = Ab + (size_t)arow * K + akoff;
  for (int kb = 0; kb < K; kb += 32) {
    v16bf a = ld_a_f32(pa + kb);
    if (kb + 32 < K) __builtin_prefetch(pa + kb + 32, 0, 1);   // global_prefetch_b8
#pragma unroll
    for (int nt = 0; nt < 8; ++nt) {
      v16bf bf = ld_b(Bb + (size_t)(nt * 16 + bn) * K + kb + bkoff);
      acc[nt] = WMMA_BF16(a, bf, acc[nt]);
    }
  }
  float* Ob = Out + ((size_t)pb * SEQ + st * 128) * D_REPR;
#pragma unroll
  for (int nt = 0; nt < 8; ++nt) st_c(Ob, D_REPR, m, nt * 16, lane, acc[nt], bs);
}

// ---------------------------------------------------------------------------
// Mean over seq (two-stage, deterministic): Mc = mean_s(C); reps0 = Mc
// ---------------------------------------------------------------------------
__global__ __launch_bounds__(128) void pcn_mean_partial(const float* __restrict__ C,
                                                        float* __restrict__ partial) {
  int ch = blockIdx.x, b = blockIdx.y, p = blockIdx.z;
  int n = threadIdx.x;
  size_t base = (((size_t)p * BATCH + b) * SEQ + (size_t)ch * 128) * D_REPR;
  float s = 0.f;
  for (int r = 0; r < 128; ++r) s += C[base + (size_t)r * D_REPR + n];
  partial[((size_t)(p * BATCH + b) * 16 + ch) * D_REPR + n] = s;
}

__global__ __launch_bounds__(128) void pcn_mean_finish(const float* __restrict__ partial,
                                                       float* __restrict__ Mc,
                                                       float* __restrict__ reps) {
  int pb = blockIdx.x;  // 16
  int n = threadIdx.x;
  float s = 0.f;
  for (int ch = 0; ch < 16; ++ch) s += partial[((size_t)pb * 16 + ch) * D_REPR + n];
  float m = s * (1.0f / SEQ);
  Mc[(size_t)pb * D_REPR + n] = m;
  reps[(size_t)pb * D_REPR + n] = m;
}

// ---------------------------------------------------------------------------
// pred[i] = reps[i+1]@Wp[i+1] + bp[i+1];  predWb[i] = pred[i]@Wprec[i][128:,:]
// grid: 12 = (i in 0..2) x batch, 128 threads
// ---------------------------------------------------------------------------
__global__ __launch_bounds__(128) void pcn_pred(const float* __restrict__ reps,
                                                const float* __restrict__ Wp,
                                                const float* __restrict__ bp,
                                                const float* __restrict__ Wprec,
                                                float* __restrict__ pred,
                                                float* __restrict__ predWb) {
  int i = blockIdx.x >> 2, b = blockIdx.x & 3;
  int n = threadIdx.x;
  __shared__ float sp[D_REPR];
  const float* r = reps + ((size_t)(i + 1) * BATCH + b) * D_REPR;
  const float* W = Wp + (size_t)(i + 1) * D_REPR * D_REPR;
  float acc = bp[(i + 1) * D_REPR + n];
  for (int k = 0; k < D_REPR; ++k) acc = fmaf(r[k], W[(size_t)k * D_REPR + n], acc);
  int pbo = (i * BATCH + b) * D_REPR;
  pred[pbo + n] = acc;
  sp[n] = acc;
  __syncthreads();
  const float* Wb = Wprec + (size_t)i * 2 * D_REPR * D_REPR + (size_t)D_REPR * D_REPR;
  float a2 = 0.f;
  for (int k = 0; k < D_REPR; ++k) a2 = fmaf(sp[k], Wb[(size_t)k * D_REPR + n], a2);
  predWb[pbo + n] = a2;
}

// ---------------------------------------------------------------------------
// Settle update: rep[i] = pred + mean_s((C-pred)*sigmoid(Zc-predWb)), i in 0..2
// ---------------------------------------------------------------------------
__global__ __launch_bounds__(256) void pcn_settle_partial(const float* __restrict__ C,
                                                          const float* __restrict__ Zc,
                                                          const float* __restrict__ pred,
                                                          const float* __restrict__ predWb,
                                                          float* __restrict__ partial) {
  int ch = blockIdx.x, b = blockIdx.y, i = blockIdx.z;   // 16 x 4 x 3
  int n = threadIdx.x & 127, half = threadIdx.x >> 7;
  size_t base = (((size_t)i * BATCH + b) * SEQ + (size_t)ch * 128) * D_REPR;
  int pbo = (i * BATCH + b) * D_REPR;
  float pv = pred[pbo + n];
  float pw = predWb[pbo + n];
  float s = 0.f;
  for (int r = half; r < 128; r += 2) {
    float c = C[base + (size_t)r * D_REPR + n];
    float z = Zc[base + (size_t)r * D_REPR + n];
    float sg = 1.f / (1.f + __expf(-(z - pw)));
    s += (c - pv) * sg;
  }
  __shared__ float red[256];
  red[threadIdx.x] = s;
  __syncthreads();
  if (half == 0)
    partial[((size_t)(i * BATCH + b) * 16 + ch) * D_REPR + n] = red[n] + red[n + 128];
}

__global__ __launch_bounds__(128) void pcn_settle_finish(const float* __restrict__ partial,
                                                         const float* __restrict__ pred,
                                                         float* __restrict__ reps) {
  int i = blockIdx.x >> 2, b = blockIdx.x & 3;  // 12
  int n = threadIdx.x;
  int pbo = (i * BATCH + b) * D_REPR;
  float s = 0.f;
  for (int ch = 0; ch < 16; ++ch) s += partial[((size_t)(i * BATCH + b) * 16 + ch) * D_REPR + n];
  reps[pbo + n] = pred[pbo + n] + s * (1.0f / SEQ);
}

// ---------------------------------------------------------------------------
// loss = (1/3) * sum_{i<3} mean((pred_i - Mc_i)^2)   (mean over 4*128 = 512)
// ---------------------------------------------------------------------------
__global__ __launch_bounds__(256) void pcn_loss(const float* __restrict__ pred,
                                                const float* __restrict__ Mc,
                                                float* __restrict__ out) {
  float s = 0.f;
  for (int idx = threadIdx.x; idx < 3 * BATCH * D_REPR; idx += 256) {
    float d = pred[idx] - Mc[idx];
    s += d * d;
  }
  __shared__ float red[256];
  red[threadIdx.x] = s;
  for (int stp = 128; stp > 0; stp >>= 1) {
    __syncthreads();
    if (threadIdx.x < stp) red[threadIdx.x] += red[threadIdx.x + stp];
  }
  if (threadIdx.x == 0) out[0] = red[0] * (1.0f / (3.0f * BATCH * D_REPR));
}

// ---------------------------------------------------------------------------
// we = (C - pred) * sigmoid(Zc - predWb), stored bf16 (A operand of corr GEMM)
// patch 3: pred = predWb = 0 (matches reference: err=c, concat(c,c)@Wprec = Zc)
// ---------------------------------------------------------------------------
__global__ __launch_bounds__(256) void pcn_we(const float* __restrict__ C,
                                              const float* __restrict__ Zc,
                                              const float* __restrict__ pred,
                                              const float* __restrict__ predWb,
                                              __bf16* __restrict__ we) {
  size_t idx = (size_t)blockIdx.x * 256 + threadIdx.x;  // 4*4*2048*128
  int n = (int)(idx & 127);
  size_t r = idx >> 7;
  int pb = (int)(r >> 11);
  float c = C[idx], z = Zc[idx];
  float pv = pred[pb * D_REPR + n];
  float pw = predWb[pb * D_REPR + n];
  float sg = 1.f / (1.f + __expf(-(z - pw)));
  we[idx] = (__bf16)((c - pv) * sg);
}

// ---------------------------------------------------------------------------
// corrections = we @ Wcorr + bcorr   (M-tile 128, N-tile 128 of 2048, K=128)
// grid: (ntile=16, seqtile=16, pb=16)
// ---------------------------------------------------------------------------
__global__ __launch_bounds__(256) void pcn_corr(const __bf16* __restrict__ we,
                                                const __bf16* __restrict__ WcorrT,
                                                const float* __restrict__ bcorr,
                                                float* __restrict__ Out) {
  int ntb = blockIdx.x, st = blockIdx.y, pb = blockIdx.z;
  int p = pb >> 2;
  int lane = threadIdx.x & 31, wave = threadIdx.x >> 5;
  const __bf16* Ab = we + ((size_t)pb * SEQ + st * 128) * D_REPR;
  const __bf16* Bb = WcorrT + ((size_t)p * D_MODEL + (size_t)ntb * 128) * D_REPR;
  const float*  bs = bcorr + (size_t)p * D_MODEL + (size_t)ntb * 128;

  int m = wave * 16;
  int arow  = m + (lane & 15);
  int akoff = (lane & 16) ? 8 : 0;
  int bn    = lane & 15;
  int bkoff = (lane & 16) ? 8 : 0;

  v8f acc[8] = {};
  const __bf16* pa = Ab + (size_t)arow * D_REPR + akoff;
#pragma unroll
  for (int kb = 0; kb < D_REPR; kb += 32) {
    v16bf a = ld_a_bf(pa + kb);
#pragma unroll
    for (int nt = 0; nt < 8; ++nt) {
      v16bf bf = ld_b(Bb + (size_t)(nt * 16 + bn) * D_REPR + kb + bkoff);
      acc[nt] = WMMA_BF16(a, bf, acc[nt]);
    }
  }
  float* Ob = Out + ((size_t)pb * SEQ + st * 128) * D_MODEL + (size_t)ntb * 128;
#pragma unroll
  for (int nt = 0; nt < 8; ++nt) st_c(Ob, D_MODEL, m, nt * 16, lane, acc[nt], bs);
}

// ---------------------------------------------------------------------------
// Launch
// ---------------------------------------------------------------------------
static inline size_t align_up(size_t x) { return (x + 255) & ~(size_t)255; }

extern "C" void kernel_launch(void* const* d_in, const int* in_sizes, int n_in,
                              void* d_out, int out_size, void* d_ws, size_t ws_size,
                              hipStream_t stream) {
  (void)in_sizes; (void)n_in; (void)out_size; (void)ws_size;
  const float* h     = (const float*)d_in[0];
  const float* Wc    = (const float*)d_in[1];
  const float* bc    = (const float*)d_in[2];
  const float* Wp    = (const float*)d_in[3];
  const float* bp    = (const float*)d_in[4];
  const float* Wprec = (const float*)d_in[5];
  const float* bprec = (const float*)d_in[6];
  const float* Wcorr = (const float*)d_in[7];
  const float* bcorr = (const float*)d_in[8];
  float* out = (float*)d_out;

  // workspace carve-up
  char* cur = (char*)d_ws;
  const size_t nBSR = (size_t)N_PATCH * BATCH * SEQ * D_REPR;          // 4,194,304
  float*  C       = (float*)cur;  cur += align_up(nBSR * 4);
  float*  Zc      = (float*)cur;  cur += align_up(nBSR * 4);
  __bf16* we      = (__bf16*)cur; cur += align_up(nBSR * 2);
  __bf16* WcT     = (__bf16*)cur; cur += align_up((size_t)N_PATCH * D_REPR * D_MODEL * 2);
  __bf16* WcorrT  = (__bf16*)cur; cur += align_up((size_t)N_PATCH * D_MODEL * D_REPR * 2);
  __bf16* WsumT   = (__bf16*)cur; cur += align_up((size_t)N_PATCH * D_REPR * D_REPR * 2);
  float*  Mc      = (float*)cur;  cur += align_up((size_t)N_PATCH * BATCH * D_REPR * 4);
  float*  reps    = (float*)cur;  cur += align_up((size_t)N_PATCH * BATCH * D_REPR * 4);
  float*  pred    = (float*)cur;  cur += align_up((size_t)N_PATCH * BATCH * D_REPR * 4);
  float*  predWb  = (float*)cur;  cur += align_up((size_t)N_PATCH * BATCH * D_REPR * 4);
  float*  partial = (float*)cur;  cur += align_up((size_t)N_PATCH * BATCH * 16 * D_REPR * 4);

  // weight prep (bf16 [n][k] transposes) + zero pred/predWb (patch 3 stays 0)
  pcn_prep_wct  <<<4096, 256, 0, stream>>>(Wc, WcT);
  pcn_prep_wsum <<<256,  256, 0, stream>>>(Wprec, WsumT);
  pcn_prep_wcorr<<<4096, 256, 0, stream>>>(Wcorr, WcorrT);
  pcn_zero_vec  <<<8,    256, 0, stream>>>(pred, predWb);

  // C = h @ Wc + bc   (K = 2048)
  pcn_gemm_n128<<<dim3(16, BATCH, N_PATCH), 256, 0, stream>>>(h, WcT, bc, C, D_MODEL);

  // Mc = mean_s(C); reps = Mc
  pcn_mean_partial<<<dim3(16, BATCH, N_PATCH), 128, 0, stream>>>(C, partial);
  pcn_mean_finish <<<N_PATCH * BATCH, 128, 0, stream>>>(partial, Mc, reps);

  // Zc = C @ (Wprec_top + Wprec_bot) + bprec   (K = 128)
  pcn_gemm_n128<<<dim3(16, BATCH, N_PATCH), 256, 0, stream>>>(C, WsumT, bprec, Zc, D_REPR);

  // settle iterations
  for (int it = 0; it < N_SETTLE; ++it) {
    pcn_pred          <<<12, 128, 0, stream>>>(reps, Wp, bp, Wprec, pred, predWb);
    pcn_settle_partial<<<dim3(16, BATCH, 3), 256, 0, stream>>>(C, Zc, pred, predWb, partial);
    pcn_settle_finish <<<12, 128, 0, stream>>>(partial, pred, reps);
  }

  // final top-down predictions, loss, weighted errors, corrections
  pcn_pred<<<12, 128, 0, stream>>>(reps, Wp, bp, Wprec, pred, predWb);
  pcn_loss<<<1, 256, 0, stream>>>(pred, Mc, out);
  pcn_we  <<<(int)(nBSR / 256), 256, 0, stream>>>(C, Zc, pred, predWb, we);
  pcn_corr<<<dim3(16, 16, N_PATCH * BATCH), 256, 0, stream>>>(we, WcorrT, bcorr, out + 1);
}
